// GNNLayer_53291954209278
// MI455X (gfx1250) — matrix operand order repaired and checked by reading, not compiled, not run on hardware
//
#include <hip/hip_runtime.h>
#include <hip/hip_bf16.h>

// ---------------------------------------------------------------------------
// GNN layer: GCNConv (sym-norm, self-loops) + BatchNorm(train) + ReLU + residual
// N = 100000 nodes, d = 128 features, E = 1.6M edges.
//
// MI455X reasoning: runtime is bound by the ~1.6 GB irregular edge
// gather/scatter (hot in the 192MB L2); the 3.3 GFLOP GEMM runs on
// v_wmma_f32_16x16x32_bf16 (bf16 in, f32 accumulate), 32-row tiles per block
// so each B fragment is reused by two accumulators. wave32 everywhere.
// ---------------------------------------------------------------------------

typedef __attribute__((ext_vector_type(16))) __bf16 v16bf;
typedef __attribute__((ext_vector_type(8)))  float  v8f;

#define DIM 128
#define LDS_PITCH 136   // 272B row stride: conflict-free ds_load_b128, 16B aligned

// ---- degree init (self-loop = 1) + zero BN stats --------------------------
__global__ void k_deg_init(float* __restrict__ deg, float* __restrict__ stats, int N) {
    int i = blockIdx.x * blockDim.x + threadIdx.x;
    if (i < N) deg[i] = 1.0f;
    if (i < 256) stats[i] = 0.0f;
}

// ---- degree accumulate over edge destinations -----------------------------
__global__ void k_deg_edges(const int* __restrict__ dst, float* __restrict__ deg, int E) {
    int e = blockIdx.x * blockDim.x + threadIdx.x;
    if (e >= E) return;
    __hip_atomic_fetch_add(&deg[dst[e]], 1.0f, __ATOMIC_RELAXED, __HIP_MEMORY_SCOPE_AGENT);
}

// ---- deg -> deg^{-1/2} in place -------------------------------------------
__global__ void k_rsqrt(float* __restrict__ deg, int N) {
    int i = blockIdx.x * blockDim.x + threadIdx.x;
    if (i >= N) return;
    float d = deg[i];
    deg[i] = (d > 0.0f) ? rsqrtf(d) : 0.0f;
}

// ---- W (f32, row-major [j][k]) -> bf16 ------------------------------------
__global__ void k_convert_w(const float* __restrict__ W, unsigned short* __restrict__ Wb) {
    int i = blockIdx.x * blockDim.x + threadIdx.x;  // 128*128 = 16384
    if (i >= DIM * DIM) return;
    ((__bf16*)Wb)[i] = (__bf16)W[i];
}

// ---- h = x @ W^T via WMMA bf16 --------------------------------------------
// Block = 256 threads = 8 waves, handles a 32-row tile of x. Wave w produces
// the 32x16 output tile for columns [16w, 16w+16) as two 16x16 accumulators
// sharing every B fragment: 8 x v_wmma_f32_16x16x32_bf16 per wave.
__global__ void __launch_bounds__(256)
k_gemm(const float* __restrict__ x, const unsigned short* __restrict__ Wb_,
       float* __restrict__ h, int N) {
    const __bf16* Wb = (const __bf16*)Wb_;
    __shared__ __bf16 ldsx[32][LDS_PITCH];

    const int tid     = threadIdx.x;
    const int rowBase = blockIdx.x * 32;

    // Stage the 32x128 x tile into LDS as bf16 (16 elements / thread).
    {
        int e0 = tid * 16;
        int r  = e0 >> 7;          // 0..31
        int c  = e0 & (DIM - 1);   // 0 or 64... actually 0..127 step 16
        int rr = rowBase + r; if (rr > N - 1) rr = N - 1;   // clamp tail reads
        const float* src = x + (size_t)rr * DIM + c;
        float4 f0 = *(const float4*)(src);
        float4 f1 = *(const float4*)(src + 4);
        float4 f2 = *(const float4*)(src + 8);
        float4 f3 = *(const float4*)(src + 12);
        __bf16* dp = &ldsx[r][c];
        dp[0]  = (__bf16)f0.x; dp[1]  = (__bf16)f0.y; dp[2]  = (__bf16)f0.z; dp[3]  = (__bf16)f0.w;
        dp[4]  = (__bf16)f1.x; dp[5]  = (__bf16)f1.y; dp[6]  = (__bf16)f1.z; dp[7]  = (__bf16)f1.w;
        dp[8]  = (__bf16)f2.x; dp[9]  = (__bf16)f2.y; dp[10] = (__bf16)f2.z; dp[11] = (__bf16)f2.w;
        dp[12] = (__bf16)f3.x; dp[13] = (__bf16)f3.y; dp[14] = (__bf16)f3.z; dp[15] = (__bf16)f3.w;
    }
    __syncthreads();

    const int wave    = tid >> 5;       // 0..7 -> output column tile
    const int lane    = tid & 31;
    const int hh      = lane >> 4;      // lane half
    const int lm      = lane & 15;
    const int colBase = wave * 16;

    v8f acc0 = {}, acc1 = {};
    #pragma unroll
    for (int kc = 0; kc < DIM; kc += 32) {
        // B fragment (32x16 bf16): column = colBase+lm, K = kc+16*hh..+15,
        // contiguous in row-major W[j][k] -> single 32B load; reused twice.
        v16bf b = *(const v16bf*)&Wb[(size_t)(colBase + lm) * DIM + kc + 16 * hh];

        // A fragments (16x32 bf16, contiguous K to match B): lane half hh holds
        // K = kc+16*hh..+15 for row lm (+16 for 2nd tile).
        const __bf16* a0lo = &ldsx[lm][kc + 16 * hh];
        const __bf16* a0hi = &ldsx[lm][kc + 16 * hh + 8];
        const __bf16* a1lo = &ldsx[lm + 16][kc + 16 * hh];
        const __bf16* a1hi = &ldsx[lm + 16][kc + 16 * hh + 8];
        v16bf a0, a1;
        #pragma unroll
        for (int i = 0; i < 8; ++i) {
            a0[i] = a0lo[i]; a0[i + 8] = a0hi[i];
            a1[i] = a1lo[i]; a1[i + 8] = a1hi[i];
        }

        acc0 = __builtin_amdgcn_wmma_f32_16x16x32_bf16(
            false, a0, false, b, (short)0, acc0, false, false);
        acc1 = __builtin_amdgcn_wmma_f32_16x16x32_bf16(
            false, a1, false, b, (short)0, acc1, false, false);
    }

    // C/D layout: VGPR r -> row r + 8*hh, lane&15 -> column.
    float* hp = h + (size_t)rowBase * DIM + colBase + lm;
    if (rowBase + 32 <= N) {            // fast path: full tile, no branches
        #pragma unroll
        for (int r = 0; r < 8; ++r) {
            hp[(size_t)(r + 8 * hh) * DIM]      = acc0[r];
            hp[(size_t)(r + 8 * hh + 16) * DIM] = acc1[r];
        }
    } else {                            // rare ragged tail
        #pragma unroll
        for (int r = 0; r < 8; ++r) {
            int r0 = rowBase + r + 8 * hh;
            int r1 = r0 + 16;
            if (r0 < N) h[(size_t)r0 * DIM + colBase + lm] = acc0[r];
            if (r1 < N) h[(size_t)r1 * DIM + colBase + lm] = acc1[r];
        }
    }
}

// ---- out = h * dinv^2 (self-loop message) + bias --------------------------
__global__ void k_init_out(const float* __restrict__ h, const float* __restrict__ dinv,
                           const float* __restrict__ bias, float* __restrict__ out,
                           long long total) {
    long long i = (long long)blockIdx.x * blockDim.x + threadIdx.x;
    if (i >= total) return;
    int node = (int)(i >> 7);
    int f    = (int)(i & (DIM - 1));
    float di = dinv[node];
    out[i] = h[i] * di * di + bias[f];
}

// ---- scatter-add edge messages: one wave per edge, float4 per lane --------
__global__ void k_edge_scatter(const int* __restrict__ src, const int* __restrict__ dst,
                               const float* __restrict__ h, const float* __restrict__ dinv,
                               float* __restrict__ out, int E) {
    long long gt = (long long)blockIdx.x * blockDim.x + threadIdx.x;
    int e = (int)(gt >> 5);             // wave-uniform
    if (e >= E) return;
    int lane = (int)(gt & 31);
    int s = src[e];
    int t = dst[e];
    float w = dinv[s] * dinv[t];
    const float4 v = *(const float4*)(h + (size_t)s * DIM + lane * 4);
    float* o = out + (size_t)t * DIM + lane * 4;
    __hip_atomic_fetch_add(o + 0, v.x * w, __ATOMIC_RELAXED, __HIP_MEMORY_SCOPE_AGENT);
    __hip_atomic_fetch_add(o + 1, v.y * w, __ATOMIC_RELAXED, __HIP_MEMORY_SCOPE_AGENT);
    __hip_atomic_fetch_add(o + 2, v.z * w, __ATOMIC_RELAXED, __HIP_MEMORY_SCOPE_AGENT);
    __hip_atomic_fetch_add(o + 3, v.w * w, __ATOMIC_RELAXED, __HIP_MEMORY_SCOPE_AGENT);
}

// ---- per-feature sum / sum-of-squares (thread f, 256 rows per block) ------
#define STAT_ROWS 256
__global__ void k_stats(const float* __restrict__ out, float* __restrict__ stats, int N) {
    int f  = threadIdx.x;                      // 0..127
    int r0 = blockIdx.x * STAT_ROWS;
    int r1 = r0 + STAT_ROWS; if (r1 > N) r1 = N;
    float s = 0.0f, ss = 0.0f;
    for (int r = r0; r < r1; ++r) {
        float v = out[(size_t)r * DIM + f];
        s += v; ss += v * v;
    }
    __hip_atomic_fetch_add(&stats[f],       s,  __ATOMIC_RELAXED, __HIP_MEMORY_SCOPE_AGENT);
    __hip_atomic_fetch_add(&stats[DIM + f], ss, __ATOMIC_RELAXED, __HIP_MEMORY_SCOPE_AGENT);
}

// ---- BN(train) + ReLU + residual, in place on out -------------------------
__global__ void k_apply(float* __restrict__ out, const float* __restrict__ x,
                        const float* __restrict__ gamma, const float* __restrict__ beta,
                        const float* __restrict__ stats, long long total, float invN) {
    long long i = (long long)blockIdx.x * blockDim.x + threadIdx.x;
    if (i >= total) return;
    int f = (int)(i & (DIM - 1));
    float mu  = stats[f] * invN;
    float var = stats[DIM + f] * invN - mu * mu;
    float v = out[i];
    v = gamma[f] * (v - mu) * rsqrtf(var + 1e-5f) + beta[f];
    v = fmaxf(v, 0.0f) + x[i];
    out[i] = v;
}

extern "C" void kernel_launch(void* const* d_in, const int* in_sizes, int n_in,
                              void* d_out, int out_size, void* d_ws, size_t ws_size,
                              hipStream_t stream) {
    const float* x     = (const float*)d_in[0];
    const float* W     = (const float*)d_in[1];
    const float* bias  = (const float*)d_in[2];
    const float* gamma = (const float*)d_in[3];
    const float* beta  = (const float*)d_in[4];
    const int*   ei    = (const int*)d_in[5];   // [2][E] flat

    float* out = (float*)d_out;
    const int N = in_sizes[0] / DIM;            // 100000
    const int E = in_sizes[5] / 2;              // 1600000
    const int* src = ei;
    const int* dst = ei + E;

    // workspace layout
    float*          h     = (float*)d_ws;                    // N*128 f32 (51.2 MB)
    float*          dinv  = h + (size_t)N * DIM;             // N f32 (deg, then deg^-1/2)
    float*          stats = dinv + N;                        // 256 f32 (sums | sumsq)
    unsigned short* Wb    = (unsigned short*)(stats + 256);  // 128*128 bf16

    const long long total = (long long)N * DIM;

    k_convert_w   <<<(DIM * DIM + 255) / 256, 256, 0, stream>>>(W, Wb);
    k_deg_init    <<<(N + 255) / 256, 256, 0, stream>>>(dinv, stats, N);
    k_deg_edges   <<<(E + 255) / 256, 256, 0, stream>>>(dst, dinv, E);
    k_rsqrt       <<<(N + 255) / 256, 256, 0, stream>>>(dinv, N);
    k_gemm        <<<(N + 31) / 32, 256, 0, stream>>>(x, Wb, h, N);
    k_init_out    <<<(int)((total + 255) / 256), 256, 0, stream>>>(h, dinv, bias, out, total);
    k_edge_scatter<<<(int)(((long long)E * 32 + 255) / 256), 256, 0, stream>>>(src, dst, h, dinv, out, E);
    k_stats       <<<(N + STAT_ROWS - 1) / STAT_ROWS, DIM, 0, stream>>>(out, stats, N);
    k_apply       <<<(int)((total + 255) / 256), 256, 0, stream>>>(out, x, gamma, beta, stats, total, 1.0f / (float)N);
}
